// LinearTransformerBlock_63488206569599
// MI455X (gfx1250) — matrix operand order repaired and checked by reading, not compiled, not run on hardware
//
#include <hip/hip_runtime.h>

// ---------------------------------------------------------------------------
// Problem constants (match reference)
// ---------------------------------------------------------------------------
#define BDIM   8
#define TDIM   4096
#define NROWS  (BDIM * TDIM)      // 32768
#define DDIM   1024
#define HEADS  16
#define KSZ    64
#define TCH    16                 // T-split for kv state kernel

typedef unsigned short u16;
typedef unsigned int   u32;
typedef unsigned long long u64;

typedef __bf16 v16bf __attribute__((ext_vector_type(16)));
typedef float  v8f   __attribute__((ext_vector_type(8)));
typedef u32    u32x4 __attribute__((ext_vector_type(4)));

union Frag {
    u32x4 q[2];
    v16bf v;
};

// ---------------------------------------------------------------------------
// helpers
// ---------------------------------------------------------------------------
__device__ __forceinline__ float bf2f(u16 h) {
    union { u32 u; float f; } x;
    x.u = ((u32)h) << 16;
    return x.f;
}
__device__ __forceinline__ u16 f2bf(float f) {
    union { float f; u32 u; } x;
    x.f = f;
    u32 r = x.u + 0x7fffu + ((x.u >> 16) & 1u);
    return (u16)(r >> 16);
}
__device__ __forceinline__ u32 pack_bf2(float a, float b) {
    return (u32)f2bf(a) | ((u32)f2bf(b) << 16);
}
// async global->LDS copy of 16 bytes per lane (CDNA5 ASYNCcnt path)
__device__ __forceinline__ void async_b128(u32 lds_off, const void* gaddr) {
    asm volatile("global_load_async_to_lds_b128 %0, %1, off"
                 :: "v"(lds_off), "v"((u64)gaddr) : "memory");
}

// ---------------------------------------------------------------------------
// Kernel 1: dual LayerNorm.  h = LN1(x); h2 = LN2(h).  One wave per row.
// ---------------------------------------------------------------------------
__global__ __launch_bounds__(256)
void ln_kernel(const float* __restrict__ x,
               const float* __restrict__ s1, const float* __restrict__ b1,
               const float* __restrict__ s2, const float* __restrict__ b2,
               u16* __restrict__ h_bf, u16* __restrict__ h2_bf)
{
    const int wave = threadIdx.x >> 5;
    const int lane = threadIdx.x & 31;
    const int row  = blockIdx.x * 8 + wave;
    const float* xr = x + (size_t)row * DDIM;

    float vals[32];
    float s = 0.f, ss = 0.f;
#pragma unroll
    for (int j = 0; j < 32; ++j) {
        float f = xr[lane + j * 32];
        vals[j] = f;
        s += f; ss += f * f;
    }
#pragma unroll
    for (int off = 16; off > 0; off >>= 1) {
        s  += __shfl_xor(s,  off, 32);
        ss += __shfl_xor(ss, off, 32);
    }
    const float mean = s * (1.0f / DDIM);
    const float var  = ss * (1.0f / DDIM) - mean * mean;
    const float inv  = rsqrtf(var + 1e-5f);

    float hv[32];
    float s2a = 0.f, ss2 = 0.f;
#pragma unroll
    for (int j = 0; j < 32; ++j) {
        int c = lane + j * 32;
        float h = (vals[j] - mean) * inv * s1[c] + b1[c];
        hv[j] = h;
        h_bf[(size_t)row * DDIM + c] = f2bf(h);
        s2a += h; ss2 += h * h;
    }
#pragma unroll
    for (int off = 16; off > 0; off >>= 1) {
        s2a += __shfl_xor(s2a, off, 32);
        ss2 += __shfl_xor(ss2, off, 32);
    }
    const float mean2 = s2a * (1.0f / DDIM);
    const float var2  = ss2 * (1.0f / DDIM) - mean2 * mean2;
    const float inv2  = rsqrtf(var2 + 1e-5f);
#pragma unroll
    for (int j = 0; j < 32; ++j) {
        int c = lane + j * 32;
        float h2 = (hv[j] - mean2) * inv2 * s2[c] + b2[c];
        h2_bf[(size_t)row * DDIM + c] = f2bf(h2);
    }
}

// ---------------------------------------------------------------------------
// Kernel 1b: swizzle one f32 [1024x1024] weight into bf16 fragment-major
// layout:  Wsw[(((kt*64 + nblk)*32 + lane)*8 + v)] = pack(W[k][n], W[k+1][n])
// with k = kt*32 + (lane>>4)*16 + 2v,  n = nblk*16 + (lane&15).
// After this, GEMM B-tile staging is a contiguous 8KB block copy.
// ---------------------------------------------------------------------------
__global__ __launch_bounds__(256)
void wswz_kernel(const float* __restrict__ W, u32* __restrict__ Wsw)
{
    int e    = blockIdx.x * 256 + threadIdx.x;     // 0 .. 512K-1
    int v    = e & 7;
    int l    = (e >> 3) & 31;
    int nblk = (e >> 8) & 63;
    int kt   = e >> 14;
    int k    = kt * 32 + (l >> 4) * 16 + 2 * v;
    int n    = nblk * 16 + (l & 15);
    Wsw[e] = pack_bf2(W[(size_t)k * DDIM + n], W[(size_t)(k + 1) * DDIM + n]);
}

// ---------------------------------------------------------------------------
// Kernel 2: bf16 WMMA GEMM, double-buffered async-LDS pipeline.
//   out = act(A[32768x1024] @ W + bias (+ emb[labels[row]][col%64])) (+ resid)
// Block tile 128x128, 8 waves, K-step 32.  Per k-step each thread issues
// 4x global_load_async_to_lds_b128 (loop-invariant address bases, constant
// strides); compute streams 8 v_wmma_f32_16x16x32_bf16 per wave fed by
// ds_load_b128 pairs.  s_wait_asynccnt 4 overlaps the kt+1 DMA with kt math.
// ---------------------------------------------------------------------------
__global__ __launch_bounds__(256)
void gemm_kernel(const u16* __restrict__ A,       // bf16 [NROWS x 1024]
                 const u32* __restrict__ Bsw,     // swizzled bf16 weights
                 const float* __restrict__ bias,
                 const float* __restrict__ emb,     // [NUM_TAGS x 64] | null
                 const int*   __restrict__ labels,  // [NROWS]         | null
                 const float* __restrict__ resid,   // [NROWS x 1024]  | null
                 float*       __restrict__ outF,    // f32 out | null
                 u16*         __restrict__ outB,    // bf16 out | null
                 int act)                           // 0 none, 1 elu+1, 2 gelu
{
    __shared__ __attribute__((aligned(16))) u32 lA[2 * 2048];  // [buf][mt][lane][v]
    __shared__ __attribute__((aligned(16))) u32 lB[2 * 2048];  // [buf][nt][lane][v]

    const int tid  = threadIdx.x;
    const int wave = tid >> 5;
    const int lane = tid & 31;
    const int row0 = blockIdx.y * 128;
    const int col0 = blockIdx.x * 128;

    const u32* Au = reinterpret_cast<const u32*>(A);   // bf16-pair view, 512 u32/row

    // ---- loop-invariant staging addresses (advance by constant strides) ----
    // A entries: f = tid (+256): mt=f>>6, l=(f&63)>>1, g=f&1
    const u32* gpA[2];
    u32 loA[2][2], loB[2][2];
#pragma unroll
    for (int it = 0; it < 2; ++it) {
        int f   = tid + it * 256;
        int mt  = f >> 6;
        int rem = f & 63;
        int l   = rem >> 1;
        int g   = rem & 1;
        gpA[it] = Au + (size_t)(row0 + mt * 16 + (l & 15)) * 512 + g * 4 + (l >> 4) * 8;
        loA[0][it] = (u32)(size_t)&lA[(mt * 32 + l) * 8 + g * 4];
        loA[1][it] = loA[0][it] + 2048 * 4;
        loB[0][it] = (u32)(size_t)&lB[f * 4];
        loB[1][it] = loB[0][it] + 2048 * 4;
    }
    const u32* gpB = Bsw + ((size_t)(col0 >> 4)) * 256 + tid * 4;   // + kt*16384

    auto stage = [&](int kt, int buf) {
#pragma unroll
        for (int it = 0; it < 2; ++it)
            async_b128(loA[buf][it], gpA[it] + kt * 16);
#pragma unroll
        for (int it = 0; it < 2; ++it)
            async_b128(loB[buf][it], gpB + (size_t)kt * 16384 + it * 1024);
    };

    v8f acc[8];
#pragma unroll
    for (int nt = 0; nt < 8; ++nt)
#pragma unroll
        for (int j = 0; j < 8; ++j) acc[nt][j] = 0.f;

    stage(0, 0);

    for (int kt = 0; kt < DDIM / 32; ++kt) {
        const int cur = kt & 1;
        if (kt < DDIM / 32 - 1) {
            stage(kt + 1, cur ^ 1);
            // in-order async completion: <=4 outstanding => current tile landed
            asm volatile("s_wait_asynccnt 0x4" ::: "memory");
        } else {
            asm volatile("s_wait_asynccnt 0x0" ::: "memory");
        }
        __syncthreads();

        Frag fa;
        const u32x4* pa = reinterpret_cast<const u32x4*>(
            &lA[cur * 2048 + (wave * 32 + lane) * 8]);
        fa.q[0] = pa[0];
        fa.q[1] = pa[1];
#pragma unroll
        for (int nt = 0; nt < 8; ++nt) {
            Frag fb;
            const u32x4* pb = reinterpret_cast<const u32x4*>(
                &lB[cur * 2048 + (nt * 32 + lane) * 8]);
            fb.q[0] = pb[0];
            fb.q[1] = pb[1];
            acc[nt] = __builtin_amdgcn_wmma_f32_16x16x32_bf16(
                false, fa.v, false, fb.v, (short)0, acc[nt], false, false);
        }
        __syncthreads();   // WAR: next iteration stages into buffer `cur`
    }

    // ---- fused epilogue ----
    const int half  = lane >> 4;
    const int nlane = lane & 15;
    const int mrow  = row0 + wave * 16;
#pragma unroll
    for (int nt = 0; nt < 8; ++nt) {
        int col = col0 + nt * 16 + nlane;
        float bsv = bias[col];
#pragma unroll
        for (int v = 0; v < 8; ++v) {
            int row = mrow + v + 8 * half;
            float val = acc[nt][v] + bsv;
            if (emb) val += emb[labels[row] * KSZ + (col & 63)];
            if (act == 1) {
                val = (val > 0.f) ? (val + 1.f) : __expf(val);   // elu(x)+1
            } else if (act == 2) {
                float c3 = val * val * val;
                val = 0.5f * val * (1.f + tanhf(0.7978845608f * (val + 0.044715f * c3)));
            }
            if (resid) val += resid[(size_t)row * DDIM + col];
            if (outF) outF[(size_t)row * DDIM + col] = val;
            else      outB[(size_t)row * DDIM + col] = f2bf(val);
        }
    }
}

// ---------------------------------------------------------------------------
// Kernel 3: linear-attention state partials.  Grid (128 bh, 16 t-chunks).
//   Mpart[i][j] = sum_{t in chunk} kf[b,t,h,i] * v[b,t,h,j],  kspart[i] likewise.
// fp32 accumulation; async b128 LDS staging.
// ---------------------------------------------------------------------------
__global__ __launch_bounds__(256)
void kv_kernel(const u16* __restrict__ kbf, const u16* __restrict__ vbf,
               float* __restrict__ kvpart, float* __restrict__ kspart)
{
    const int bh  = blockIdx.x;
    const int tch = blockIdx.y;
    const int b   = bh >> 4;
    const int hh  = bh & 15;

    __shared__ __attribute__((aligned(16))) u32 lk[32 * 32];  // 32 t x 64 bf16
    __shared__ __attribute__((aligned(16))) u32 lv[32 * 32];

    const int tid = threadIdx.x;
    const int kk  = tid >> 2;
    const int grp = tid & 3;

    const u32* ku = reinterpret_cast<const u32*>(kbf);
    const u32* vu = reinterpret_cast<const u32*>(vbf);

    float acc[16];
#pragma unroll
    for (int j = 0; j < 16; ++j) acc[j] = 0.f;
    float ks = 0.f;

    const int tl = tid >> 3, p4 = tid & 7;                // staging coords
    const u32 lofs_k = (u32)(size_t)&lk[tl * 32 + p4 * 4];
    const u32 lofs_v = (u32)(size_t)&lv[tl * 32 + p4 * 4];
    size_t src = (size_t)(b * TDIM + tch * (TDIM / TCH) + tl) * 512 + hh * 32 + p4 * 4;

    for (int c8 = 0; c8 < (TDIM / TCH) / 32; ++c8, src += 32 * 512) {
        __syncthreads();                                   // WAR on LDS
        async_b128(lofs_k, ku + src);
        async_b128(lofs_v, vu + src);
        asm volatile("s_wait_asynccnt 0x0" ::: "memory");
        __syncthreads();
#pragma unroll 4
        for (int t = 0; t < 32; ++t) {
            u32 kw = lk[t * 32 + (kk >> 1)];
            float kf = bf2f((kk & 1) ? (u16)(kw >> 16) : (u16)(kw & 0xffff));
            if (grp == 0) ks += kf;
            int base = t * 32 + grp * 8;
#pragma unroll
            for (int jj = 0; jj < 8; ++jj) {
                u32 vw = lv[base + jj];
                acc[2 * jj]     += kf * bf2f((u16)(vw & 0xffff));
                acc[2 * jj + 1] += kf * bf2f((u16)(vw >> 16));
            }
        }
    }

    size_t o = (((size_t)tch * 128 + bh) * 64 + kk) * 64 + grp * 16;
#pragma unroll
    for (int j = 0; j < 16; ++j) kvpart[o + j] = acc[j];
    if (grp == 0) kspart[((size_t)tch * 128 + bh) * 64 + kk] = ks;
}

// Deterministic reduction of the 16 T-chunk partials.
__global__ __launch_bounds__(256)
void kvred_kernel(const float* __restrict__ kvpart, const float* __restrict__ kspart,
                  float* __restrict__ kvmat, float* __restrict__ ksum)
{
    int e = blockIdx.x * 256 + threadIdx.x;        // 0 .. 524287
    float s = 0.f;
#pragma unroll
    for (int c = 0; c < TCH; ++c) s += kvpart[(size_t)c * 524288 + e];
    kvmat[e] = s;
    if (e < 8192) {
        float t = 0.f;
#pragma unroll
        for (int c = 0; c < TCH; ++c) t += kspart[(size_t)c * 8192 + e];
        ksum[e] = t;
    }
}

// ---------------------------------------------------------------------------
// Kernel 4: attn[b,t,h,d] = (sum_c q[c] * M[d][c]) / (sum_c q[c] * ksum[c])
// ---------------------------------------------------------------------------
__global__ __launch_bounds__(256)
void attn_kernel(const u16* __restrict__ qbf, const float* __restrict__ kvmat,
                 const float* __restrict__ ksum, u16* __restrict__ attnbf)
{
    const int bh = blockIdx.y;
    const int b  = bh >> 4;
    const int hh = bh & 15;
    const int t0 = blockIdx.x * 64;

    __shared__ float Ms[64 * 64];   // Ms[c*64 + d] = M[d][c]  (transposed)
    __shared__ float kss[64];
    __shared__ u32   lq[64 * 32];   // 64 t-rows x 64 bf16

    const int tid = threadIdx.x;
    const u32* qu = reinterpret_cast<const u32*>(qbf);

#pragma unroll
    for (int it = 0; it < 16; ++it) {
        int i2 = tid + it * 256;
        Ms[i2] = kvmat[(size_t)bh * 4096 + (size_t)(i2 & 63) * 64 + (i2 >> 6)];
    }
    if (tid < 64) kss[tid] = ksum[bh * 64 + tid];
#pragma unroll
    for (int it = 0; it < 8; ++it) {
        int e  = tid + it * 256;
        int tl = e >> 5;
        int p  = e & 31;
        lq[e] = qu[(size_t)(b * TDIM + t0 + tl) * 512 + hh * 32 + p];
    }
    __syncthreads();

    const int tl  = tid >> 2;
    const int grp = tid & 3;

    float q[64];
#pragma unroll
    for (int p = 0; p < 32; ++p) {
        u32 w = lq[tl * 32 + p];
        q[2 * p]     = bf2f((u16)(w & 0xffff));
        q[2 * p + 1] = bf2f((u16)(w >> 16));
    }
    float norm = 0.f;
#pragma unroll
    for (int c = 0; c < 64; ++c) norm += q[c] * kss[c];
    const float invn = 1.f / norm;

    size_t orow = (size_t)(b * TDIM + t0 + tl) * DDIM + hh * 64;
#pragma unroll
    for (int j = 0; j < 16; ++j) {
        int d = grp * 16 + j;
        float a = 0.f;
#pragma unroll
        for (int c = 0; c < 64; ++c) a += q[c] * Ms[c * 64 + d];
        attnbf[orow + d] = f2bf(a * invn);
    }
}

// ---------------------------------------------------------------------------
// Launch
// ---------------------------------------------------------------------------
extern "C" void kernel_launch(void* const* d_in, const int* in_sizes, int n_in,
                              void* d_out, int out_size, void* d_ws, size_t ws_size,
                              hipStream_t stream) {
    (void)in_sizes; (void)n_in; (void)out_size; (void)ws_size;

    const float* x     = (const float*)d_in[0];
    const int*   lab   = (const int*)  d_in[1];
    const float* emb_k = (const float*)d_in[2];
    const float* emb_v = (const float*)d_in[3];
    const float* ln1_s = (const float*)d_in[4];
    const float* ln1_b = (const float*)d_in[5];
    const float* wq    = (const float*)d_in[6];
    const float* bq    = (const float*)d_in[7];
    const float* wk    = (const float*)d_in[8];
    const float* bk    = (const float*)d_in[9];
    const float* wv    = (const float*)d_in[10];
    const float* bv    = (const float*)d_in[11];
    const float* wo    = (const float*)d_in[12];
    const float* bo    = (const float*)d_in[13];
    const float* ln2_s = (const float*)d_in[14];
    const float* ln2_b = (const float*)d_in[15];
    const float* w1    = (const float*)d_in[16];
    const float* b1    = (const float*)d_in[17];
    const float* w2    = (const float*)d_in[18];
    const float* b2    = (const float*)d_in[19];

    const size_t NE = (size_t)NROWS * DDIM;            // 33.5M elements
    char* p = (char*)d_ws;
    auto alloc = [&](size_t bytes) { char* r = p; p += (bytes + 255) & ~(size_t)255; return r; };

    u16*   h_bf   = (u16*)alloc(NE * 2);
    u16*   h2_bf  = (u16*)alloc(NE * 2);
    u16*   q_bf   = (u16*)alloc(NE * 2);               // reused as gelu-out
    u16*   k_bf   = (u16*)alloc(NE * 2);               // reused as attn
    u16*   v_bf   = (u16*)alloc(NE * 2);
    float* x2     = (float*)alloc(NE * 4);
    float* kvmat  = (float*)alloc((size_t)128 * 4096 * 4);
    float* ksumb  = (float*)alloc((size_t)128 * 64 * 4);
    float* kvpart = (float*)alloc((size_t)TCH * 128 * 4096 * 4);
    float* kspart = (float*)alloc((size_t)TCH * 128 * 64 * 4);
    u32*   wq_sw  = (u32*)alloc((size_t)512 * 1024 * 4);
    u32*   wk_sw  = (u32*)alloc((size_t)512 * 1024 * 4);
    u32*   wv_sw  = (u32*)alloc((size_t)512 * 1024 * 4);
    u32*   wo_sw  = (u32*)alloc((size_t)512 * 1024 * 4);
    u32*   w1_sw  = (u32*)alloc((size_t)512 * 1024 * 4);
    u32*   w2_sw  = (u32*)alloc((size_t)512 * 1024 * 4);
    u16*   attnb  = k_bf;                              // alias: k dead after kv_kernel
    u16*   g_bf   = q_bf;                              // alias: q dead after attn_kernel

    const dim3 gemm_grid(DDIM / 128, NROWS / 128, 1);  // (8, 256)

    // 1. h = LN1(x); h2 = LN2(h);  swizzle all six weight matrices to bf16
    ln_kernel<<<NROWS / 8, 256, 0, stream>>>(x, ln1_s, ln1_b, ln2_s, ln2_b, h_bf, h2_bf);
    wswz_kernel<<<2048, 256, 0, stream>>>(wq, wq_sw);
    wswz_kernel<<<2048, 256, 0, stream>>>(wk, wk_sw);
    wswz_kernel<<<2048, 256, 0, stream>>>(wv, wv_sw);
    wswz_kernel<<<2048, 256, 0, stream>>>(wo, wo_sw);
    wswz_kernel<<<2048, 256, 0, stream>>>(w1, w1_sw);
    wswz_kernel<<<2048, 256, 0, stream>>>(w2, w2_sw);
    // 2-4. q/k/v projections (fused bias, arc-embedding gather, elu+1)
    gemm_kernel<<<gemm_grid, 256, 0, stream>>>(h_bf, wq_sw, bq, nullptr, nullptr, nullptr,
                                               nullptr, q_bf, 1);
    gemm_kernel<<<gemm_grid, 256, 0, stream>>>(h_bf, wk_sw, bk, emb_k, lab, nullptr,
                                               nullptr, k_bf, 1);
    gemm_kernel<<<gemm_grid, 256, 0, stream>>>(h_bf, wv_sw, bv, emb_v, lab, nullptr,
                                               nullptr, v_bf, 0);
    // 5. kv state partials + deterministic reduction
    kv_kernel<<<dim3(128, TCH, 1), 256, 0, stream>>>(k_bf, v_bf, kvpart, kspart);
    kvred_kernel<<<2048, 256, 0, stream>>>(kvpart, kspart, kvmat, ksumb);
    // 6. normalized attention outputs
    attn_kernel<<<dim3(TDIM / 64, 128, 1), 256, 0, stream>>>(q_bf, kvmat, ksumb, attnb);
    // 7. output projection + x residual -> x2 (f32)
    gemm_kernel<<<gemm_grid, 256, 0, stream>>>(attnb, wo_sw, bo, nullptr, nullptr, x,
                                               x2, nullptr, 0);
    // 8. FFN up + gelu
    gemm_kernel<<<gemm_grid, 256, 0, stream>>>(h2_bf, w1_sw, b1, nullptr, nullptr, nullptr,
                                               nullptr, g_bf, 2);
    // 9. FFN down + x2 residual -> final output (f32)
    gemm_kernel<<<gemm_grid, 256, 0, stream>>>(g_bf, w2_sw, b2, nullptr, nullptr, x2,
                                               (float*)d_out, nullptr, 0);
}
